// SSDLoss_36043365548628
// MI455X (gfx1250) — compile-verified, matrix operand order checked by reading, not run on hardware
//
#include <hip/hip_runtime.h>
#include <hip/hip_bf16.h>
#include <float.h>

typedef _Float16 v16h __attribute__((ext_vector_type(16)));
typedef float    v8f  __attribute__((ext_vector_type(8)));

#define NB 32        // batch
#define NA 32768     // anchors
#define NG 64        // gt boxes
#define NC 91        // classes
#define IOU_THRES 0.45f
#define XCLIP 4.135166556742356f   // log(1000/16)

// ---------- helpers ----------
__device__ __forceinline__ unsigned int fkey(float f) {
    unsigned int b = __float_as_uint(f);
    return (b & 0x80000000u) ? ~b : (b | 0x80000000u);   // monotone ascending key
}
__device__ __forceinline__ float keyval(unsigned int k) {
    unsigned int b = (k & 0x80000000u) ? (k & 0x7FFFFFFFu) : ~k;
    return __uint_as_float(b);
}

// ---------- 0: zero accumulators / per-GT best ----------
__global__ void k_init(unsigned long long* gtbest, int* fgcnt, float* accum) {
    int i = blockIdx.x * blockDim.x + threadIdx.x;
    if (i < NB * NG) gtbest[i] = 0ull;
    if (i < NB)      fgcnt[i]  = 0;
    if (i < 4)       accum[i]  = 0.0f;
}

// ---------- 1: pairwise IoU, per-anchor argmax, per-GT best anchor ----------
__global__ void k_iou_match(const float* __restrict__ anchors,
                            const float* __restrict__ gt,
                            unsigned long long* __restrict__ gtbest,
                            int* __restrict__ matched) {
    __shared__ float sgt[NG * 4];
    const int tilesPerB = NA / 256;
    const int b    = blockIdx.x / tilesPerB;
    const int tile = blockIdx.x % tilesPerB;
    const int tid  = threadIdx.x;
    if (tid < NG * 4) sgt[tid] = gt[(size_t)b * NG * 4 + tid];
    __syncthreads();

    const int a = tile * 256 + tid;
    const float* ab = anchors + ((size_t)b * NA + a) * 4;
    const float ax0 = ab[0], ay0 = ab[1], ax1 = ab[2], ay1 = ab[3];
    const float area_a = (ax1 - ax0) * (ay1 - ay0);

    float bestv = -1.0f; int bestg = 0;
    for (int g = 0; g < NG; ++g) {
        float bx0 = sgt[g*4+0], by0 = sgt[g*4+1], bx1 = sgt[g*4+2], by1 = sgt[g*4+3];
        float iw = fmaxf(fminf(bx1, ax1) - fmaxf(bx0, ax0), 0.0f);
        float ih = fmaxf(fminf(by1, ay1) - fmaxf(by0, ay0), 0.0f);
        float inter  = iw * ih;
        float area_b = (bx1 - bx0) * (by1 - by0);
        float iou = inter / (area_a + area_b - inter + 1e-16f);
        if (iou > bestv) { bestv = iou; bestg = g; }   // first-g wins ties (argmax)
        // per-GT best anchor: pack (iou_bits, ~a) so ties pick lowest anchor idx
        unsigned int hi = __float_as_uint(iou);
        unsigned int lo = ~(unsigned int)a;
        for (int s = 16; s > 0; s >>= 1) {
            unsigned int ohi = __shfl_xor(hi, s);
            unsigned int olo = __shfl_xor(lo, s);
            if (ohi > hi || (ohi == hi && olo > lo)) { hi = ohi; lo = olo; }
        }
        if ((tid & 31) == 0)
            atomicMax(&gtbest[b * NG + g], ((unsigned long long)hi << 32) | lo);
    }
    matched[(size_t)b * NA + a] = (bestv < IOU_THRES) ? -1 : bestg;
}

// ---------- 2: force-match best anchor per GT ----------
__global__ void k_scatter(const unsigned long long* __restrict__ gtbest,
                          int* __restrict__ matched) {
    int i = blockIdx.x * blockDim.x + threadIdx.x;
    if (i >= NB * NG) return;
    int b = i / NG, g = i % NG;
    unsigned int a = ~(unsigned int)(gtbest[i] & 0xFFFFFFFFu);
    matched[(size_t)b * NA + a] = g;
}

// ---------- 3: fused log-softmax CE + decode/GIoU (single pass over 382MB) ----------
// 256 threads = 8 waves; each wave owns 16 anchors (2 lanes per anchor = the two
// K-halves of the WMMA B-matrix column). Sum(exp) over classes done with
// V_WMMA_F32_16X16X32_F16 ones-matmul: D[m][n] = sum_k B[k][n].
__global__ void k_cls_giou(const float* __restrict__ logits,
                           const float* __restrict__ reg,
                           const float* __restrict__ anchors,
                           const float* __restrict__ gt,
                           const int*   __restrict__ gtlab,
                           const int*   __restrict__ matched,
                           float* __restrict__ negloss,
                           int*   __restrict__ fgcnt,
                           float* __restrict__ accum) {  // [0]=bbox_sum [1]=cls_fg_sum
    __shared__ float lds[128 * NC];                      // 46.5 KB tile
    const int tilesPerB = NA / 128;
    const int b    = blockIdx.x / tilesPerB;
    const int tile = blockIdx.x % tilesPerB;
    const int a0   = tile * 128;
    const int tid  = threadIdx.x;
    const float* src = logits + ((size_t)b * NA + a0) * NC;

    // coalesced stage of the tile into LDS; prefetch hint for the next tile
    for (int i = tid; i < 128 * NC; i += 256) lds[i] = src[i];
    if (tid == 0 && tile + 1 < tilesPerB) __builtin_prefetch(src + 128 * NC, 0, 1);
    __syncthreads();

    const int lane = tid & 31;
    const int half = lane >> 4;                  // which K-half of the column
    const int row  = ((tid >> 5) << 4) + (lane & 15);   // anchor row in tile
    const int a    = a0 + row;
    const float* lrow = &lds[row * NC];

    // per-anchor max (two lanes split the 91 classes, then exchange)
    float amax = -FLT_MAX;
    const int cbeg = half ? 46 : 0;
    const int cend = half ? NC : 46;
    for (int c = cbeg; c < cend; ++c) amax = fmaxf(amax, lrow[c]);
    amax = fmaxf(amax, __shfl_xor(amax, 16));

    // sum(exp(x - amax)) via WMMA ones-matmul, 32 classes per matrix op
    v16h ones;
    #pragma unroll
    for (int i = 0; i < 16; ++i) ones[i] = (_Float16)1.0f;
    v8f d = {0.f, 0.f, 0.f, 0.f, 0.f, 0.f, 0.f, 0.f};
    const int off = half * 8;
    for (int cb = 0; cb < NC; cb += 32) {
        v16h bm;
        #pragma unroll
        for (int i = 0; i < 8; ++i) {
            int c0 = cb + off + i;           // K = cb + {0..7 | 8..15}
            int c1 = cb + 16 + off + i;      // K = cb + {16..23 | 24..31}
            float e0 = (c0 < NC) ? __expf(lrow[c0] - amax) : 0.0f;
            float e1 = (c1 < NC) ? __expf(lrow[c1] - amax) : 0.0f;
            bm[i]     = (_Float16)e0;
            bm[8 + i] = (_Float16)e1;
        }
        d = __builtin_amdgcn_wmma_f32_16x16x32_f16(false, ones, false, bm,
                                                   (short)0, d, false, false);
    }
    const float lse = amax + __logf(d[0]);     // D[.,n] = sumexp of anchor n

    const int m = matched[(size_t)b * NA + a];
    float cls_fg = 0.0f, giou_l = 0.0f; int fgc = 0;
    if (half == 0) {                           // one lane per anchor does scalars
        const int tgt = (m >= 0) ? gtlab[b * NG + m] : (NC - 1);
        const float cls_loss = lse - lrow[tgt];
        negloss[(size_t)b * NA + a] = (m >= 0) ? -__builtin_inff() : cls_loss;
        if (m >= 0) {
            fgc = 1; cls_fg = cls_loss;
            const float* ar = anchors + ((size_t)b * NA + a) * 4;
            const float* rg = reg     + ((size_t)b * NA + a) * 4;
            float w  = ar[2] - ar[0], h = ar[3] - ar[1];
            float cx = ar[0] + 0.5f * w, cy = ar[1] + 0.5f * h;
            float dw = fminf(rg[2], XCLIP), dh = fminf(rg[3], XCLIP);
            float pcx = rg[0] * w + cx, pcy = rg[1] * h + cy;
            float pw = __expf(dw) * w, ph = __expf(dh) * h;
            float p0 = pcx - 0.5f * pw, p1 = pcy - 0.5f * ph;
            float p2 = pcx + 0.5f * pw, p3 = pcy + 0.5f * ph;
            const float* gb = gt + ((size_t)b * NG + m) * 4;
            float g0 = gb[0], g1 = gb[1], g2 = gb[2], g3 = gb[3];
            float iw = fmaxf(fminf(p2, g2) - fmaxf(p0, g0), 0.0f);
            float ih = fmaxf(fminf(p3, g3) - fmaxf(p1, g1), 0.0f);
            float inter = iw * ih;
            float ap = (p2 - p0) * (p3 - p1), ag = (g2 - g0) * (g3 - g1);
            float uni = ap + ag - inter + 1e-16f;
            float iou = inter / uni;
            float ac = (fmaxf(p2, g2) - fminf(p0, g0)) * (fmaxf(p3, g3) - fminf(p1, g1));
            float giou = iou - (ac - uni) / fmaxf(ac, 1e-16f);
            giou_l = 1.0f - fminf(fmaxf(giou, -1.0f), 1.0f);
        }
    }
    // wave reduction (lanes 16..31 carry zeros) then one atomic per wave
    for (int s = 16; s > 0; s >>= 1) {
        cls_fg += __shfl_xor(cls_fg, s);
        giou_l += __shfl_xor(giou_l, s);
        fgc    += __shfl_xor(fgc, s);
    }
    if (lane == 0) {
        if (fgc) atomicAdd(&fgcnt[b], fgc);
        atomicAdd(&accum[0], giou_l);
        atomicAdd(&accum[1], cls_fg);
    }
}

// ---------- 4: exact top-k(=3*fg) sum per batch row via 4-pass radix select ----------
__global__ void k_topk(const float* __restrict__ negloss,
                       const int*   __restrict__ fgcnt,
                       float* __restrict__ accum) {      // [2]=cls_bg_sum
    __shared__ unsigned int hist[256];
    __shared__ unsigned int s_prefix, s_remk;
    __shared__ float s_red[256];
    const int b = blockIdx.x, tid = threadIdx.x;
    const float* row = negloss + (size_t)b * NA;

    long long kk = 3LL * fgcnt[b];
    if (kk > NA) kk = NA;
    if (kk <= 0) return;                                  // uniform per block
    unsigned int prefix = 0, remk = (unsigned int)kk;

    for (int pass = 3; pass >= 0; --pass) {
        hist[tid] = 0;
        __syncthreads();
        const unsigned int himask = (pass == 3) ? 0u : (0xFFFFFFFFu << (8 * (pass + 1)));
        for (int i = tid; i < NA; i += 256) {
            unsigned int kb = fkey(row[i]);
            if ((kb & himask) == (prefix & himask))
                atomicAdd(&hist[(kb >> (8 * pass)) & 0xFF], 1u);
        }
        __syncthreads();
        if (tid == 0) {
            unsigned int cum = 0, r = remk; int sel = 0;
            for (int bin = 255; bin >= 0; --bin) {
                cum += hist[bin];
                if (cum >= r) { sel = bin; r -= (cum - hist[bin]); break; }
            }
            s_prefix = prefix | ((unsigned int)sel << (8 * pass));
            s_remk = r;
        }
        __syncthreads();
        prefix = s_prefix; remk = s_remk;
    }
    // sum of strictly-above values + tie contribution
    float lsum = 0.0f;
    for (int i = tid; i < NA; i += 256) {
        float v = row[i];
        if (fkey(v) > prefix) lsum += v;
    }
    s_red[tid] = lsum;
    __syncthreads();
    for (int s = 128; s > 0; s >>= 1) { if (tid < s) s_red[tid] += s_red[tid + s]; __syncthreads(); }
    if (tid == 0) {
        float tv = keyval(prefix);
        float total = s_red[0];
        if (remk > 0 && tv > -FLT_MAX) total += (float)remk * tv;  // skip -inf sentinel
        atomicAdd(&accum[2], total);
    }
}

// ---------- 5: finalize ----------
__global__ void k_final(const int* __restrict__ fgcnt,
                        const float* __restrict__ accum,
                        float* __restrict__ out) {
    if (threadIdx.x == 0 && blockIdx.x == 0) {
        int nfg = 0;
        for (int b = 0; b < NB; ++b) nfg += fgcnt[b];
        float N = fmaxf(1.0f, (float)nfg);
        out[0] = 2.0f * accum[0] / N;
        out[1] = (accum[1] + accum[2]) / N;
    }
}

extern "C" void kernel_launch(void* const* d_in, const int* in_sizes, int n_in,
                              void* d_out, int out_size, void* d_ws, size_t ws_size,
                              hipStream_t stream) {
    const float* cls_logits = (const float*)d_in[0];
    const float* bbox_reg   = (const float*)d_in[1];
    const float* anchors    = (const float*)d_in[2];
    const float* gt_boxes   = (const float*)d_in[3];
    const int*   gt_labels  = (const int*)d_in[4];
    float* out = (float*)d_out;

    char* w = (char*)d_ws;
    int*   matched               = (int*)(w);                               // 4 MB
    float* negloss               = (float*)(w + (size_t)NB * NA * 4);       // 4 MB
    unsigned long long* gtbest   = (unsigned long long*)(w + (size_t)NB * NA * 8);
    int*   fgcnt                 = (int*)(w + (size_t)NB * NA * 8 + NB * NG * 8);
    float* accum                 = (float*)(w + (size_t)NB * NA * 8 + NB * NG * 8 + 256);

    k_init<<<(NB * NG + 255) / 256, 256, 0, stream>>>(gtbest, fgcnt, accum);
    k_iou_match<<<NB * (NA / 256), 256, 0, stream>>>(anchors, gt_boxes, gtbest, matched);
    k_scatter<<<(NB * NG + 255) / 256, 256, 0, stream>>>(gtbest, matched);
    k_cls_giou<<<NB * (NA / 128), 256, 0, stream>>>(cls_logits, bbox_reg, anchors,
                                                    gt_boxes, gt_labels, matched,
                                                    negloss, fgcnt, accum);
    k_topk<<<NB, 256, 0, stream>>>(negloss, fgcnt, accum);
    k_final<<<1, 32, 0, stream>>>(fgcnt, accum, out);
}
